// multi_head_attn_22952305230090
// MI455X (gfx1250) — matrix operand order repaired and checked by reading, not compiled, not run on hardware
//
#include <hip/hip_runtime.h>

// Problem dims (match reference): H=16, D=1024, B=2, S=2048, DK=64
#define HH  16
#define DD  1024
#define BB  2
#define SS  2048
#define DKK 64

typedef __attribute__((ext_vector_type(16))) __bf16        v16bf;
typedef __attribute__((ext_vector_type(8)))  float         v8f;
typedef __attribute__((ext_vector_type(4)))  float         f32x4;
typedef __attribute__((ext_vector_type(4)))  unsigned int  u32x4;
typedef __attribute__((ext_vector_type(8)))  int           i32x8;
typedef __attribute__((ext_vector_type(4)))  int           i32x4;

#if defined(__has_builtin)
#  if __has_builtin(__builtin_amdgcn_tensor_load_to_lds) && \
      __has_builtin(__builtin_amdgcn_s_wait_tensorcnt)
#    define USE_TDM 1
#  endif
#  if __has_builtin(__builtin_amdgcn_global_load_async_to_lds_b128) && \
      __has_builtin(__builtin_amdgcn_s_wait_asynccnt)
#    define USE_ASYNC 1
#  endif
#endif

typedef __attribute__((address_space(1))) void as1_v;
typedef __attribute__((address_space(3))) void as3_v;

union Frag {
    v16bf  v;
    u32x4  q[2];
    __bf16 u[16];
};

__device__ __forceinline__ v8f wmma_bf16(v16bf a, v16bf b, v8f c) {
    return __builtin_amdgcn_wmma_f32_16x16x32_bf16(
        false, a, false, b, (short)0, c, false, false);
}

// A-operand (16x32) from row-major bf16: lane L holds row M=L%16,
// K-chunk [c,c+16) with c = 16*(L/16).
__device__ __forceinline__ v16bf load_a_rowmajor(const __bf16* base,
                                                 int row_stride, int k0) {
    int lane = threadIdx.x & 31;
    const __bf16* p = base + (size_t)(lane & 15) * row_stride + k0 + ((lane >> 4) * 16);
    Frag f;
    f.q[0] = *reinterpret_cast<const u32x4*>(p);
    f.q[1] = *reinterpret_cast<const u32x4*>(p + 8);
    return f.v;
}

// B-operand (32x16) from fragment-packed buffer: tile = 32 lanes x 16 bf16.
__device__ __forceinline__ v16bf load_b_packed(const __bf16* tile) {
    const __bf16* p = tile + (threadIdx.x & 31) * 16;
    Frag f;
    f.q[0] = *reinterpret_cast<const u32x4*>(p);
    f.q[1] = *reinterpret_cast<const u32x4*>(p + 8);
    return f.v;
}

// Stage a contiguous 16*DD-element bf16 strip (32 KB) from global into LDS.
// Preference: TDM tensor DMA -> async global->LDS -> plain copy.
__device__ __forceinline__ void stage_strip_32k(__bf16* lds, const __bf16* g) {
#if defined(USE_TDM)
    if ((threadIdx.x >> 5) == 0) {
        unsigned long long ga = (unsigned long long)g;
        unsigned int la = (unsigned int)(unsigned long long)lds;
        const unsigned NE = 16u * DD;                       // 16384 elements
        // D# group0: count=1 | lds_addr | global_addr | type=2
        u32x4 d0 = { 1u, la, (unsigned)(ga & 0xFFFFFFFFu),
                     (unsigned)(((ga >> 32) & 0x01FFFFFFu) | (2u << 30)) };
        // D# group1: data_size=2B; tensor_dim0=NE; tensor_dim1=1;
        //            tile_dim0=NE; tile_dim1=1; stride0=NE  (1-D contiguous tile)
        i32x8 d1 = { (int)(1u << 16),
                     (int)((NE & 0xFFFFu) << 16),
                     (int)(((NE >> 16) & 0xFFFFu) | (1u << 16)),
                     (int)((NE & 0xFFFFu) << 16),
                     (int)1,
                     (int)NE,
                     0, 0 };
        i32x4 d2 = { 0, 0, 0, 0 };
        i32x4 d3 = { 0, 0, 0, 0 };
#if defined(__clang_major__) && (__clang_major__ >= 23)
        i32x8 d4 = { 0, 0, 0, 0, 0, 0, 0, 0 };
        __builtin_amdgcn_tensor_load_to_lds(d0, d1, d2, d3, d4, 0);
#else
        __builtin_amdgcn_tensor_load_to_lds(d0, d1, d2, d3, 0);
#endif
        __builtin_amdgcn_s_wait_tensorcnt(0);
    }
    __syncthreads();
#elif defined(USE_ASYNC)
    for (int i = threadIdx.x; i < (16 * DD) / 8; i += blockDim.x) {
        __builtin_amdgcn_global_load_async_to_lds_b128(
            (as1_v*)(unsigned long long)(g + i * 8),
            (as3_v*)(unsigned int)(unsigned long long)(lds + i * 8),
            0, 0);
    }
    __builtin_amdgcn_s_wait_asynccnt(0);
    __syncthreads();
#else
    for (int i = threadIdx.x; i < (16 * DD) / 8; i += blockDim.x)
        *reinterpret_cast<u32x4*>(lds + i * 8) =
            *reinterpret_cast<const u32x4*>(g + i * 8);
    __syncthreads();
#endif
}

// ---------------------------------------------------------------- cast x -> bf16
__global__ void k_cast_bf16(const float* __restrict__ x,
                            __bf16* __restrict__ y, int n) {
    int i = blockIdx.x * blockDim.x + threadIdx.x;
    if (i < n) y[i] = (__bf16)x[i];
}

// ---------------------- pack W[h,dk,d] -> B-operand layout (K-dim=d, N-dim=dk)
__global__ void k_pack_proj(const float* __restrict__ w,
                            __bf16* __restrict__ dst) {
    int i = blockIdx.x * blockDim.x + threadIdx.x;       // over H*DK*D
    if (i >= HH * DKK * DD) return;
    int d  = i % DD;
    int dk = (i / DD) % DKK;
    int h  = i / (DD * DKK);
    int lane = (dk & 15) + (((d & 31) >> 4) << 4);
    size_t off = ((((size_t)h * (DD/32) + (d >> 5)) * (DKK/16) + (dk >> 4)) << 9)
               + (size_t)lane * 16 + (d & 15);
    dst[off] = (__bf16)w[i];
}

// ---------------- pack Wvu[h,d,kk] -> B layout with K-dim=(h*DK+kk), N-dim=d
__global__ void k_pack_w2(const float* __restrict__ wvu,
                          __bf16* __restrict__ dst) {
    int i = blockIdx.x * blockDim.x + threadIdx.x;       // over H*D*DK
    if (i >= HH * DD * DKK) return;
    int kk = i % DKK;
    int d  = (i / DKK) % DD;
    int h  = i / (DKK * DD);
    int kc = h * DKK + kk;
    int lane = (d & 15) + (((kc & 31) >> 4) << 4);
    size_t off = (((size_t)(kc >> 5) * (DD/16) + (d >> 4)) << 9)
               + (size_t)lane * 16 + (kc & 15);
    dst[off] = (__bf16)wvu[i];
}

// -------------------------------------------------------- QKV projection GEMMs
// block = 4 waves; X strip staged once in LDS; wave w owns dk cols [16w,16w+16).
__global__ void __launch_bounds__(128) k_qkv(
    const __bf16* __restrict__ xb,    // [B*S, D] bf16 row-major
    const __bf16* __restrict__ wq,    // packed B-operand
    const __bf16* __restrict__ wk,
    const __bf16* __restrict__ wv,
    __bf16* __restrict__ qout,        // [B,H,S,DK] bf16 row-major
    __bf16* __restrict__ kpk,         // packed for scores B-operand
    __bf16* __restrict__ vpk)         // packed for P*V  B-operand
{
    __shared__ __align__(16) __bf16 xs[16 * DD];          // 32 KB

    int st   = blockIdx.x & 127;
    int h    = (blockIdx.x >> 7) & 15;
    int b    = blockIdx.x >> 11;
    int wave = threadIdx.x >> 5;
    int lane = threadIdx.x & 31;
    int s0   = st * 16;

    stage_strip_32k(xs, xb + ((size_t)b * SS + s0) * DD);

    v8f cq = {}, ck = {}, cv = {};
    v16bf a  = load_a_rowmajor(xs, DD, 0);
    size_t wb = (((size_t)h * (DD/32) * (DKK/16) + wave) << 9);
    v16bf bq = load_b_packed(wq + wb);
    v16bf bk = load_b_packed(wk + wb);
    v16bf bv = load_b_packed(wv + wb);
    for (int kt = 0; kt < DD / 32 - 1; ++kt) {
        size_t wb2 = ((((size_t)h * (DD/32) + kt + 1) * (DKK/16) + wave) << 9);
        __builtin_prefetch(wq + wb2 + 2048, 0, 0);
        v16bf a2  = load_a_rowmajor(xs, DD, (kt + 1) * 32);
        v16bf bq2 = load_b_packed(wq + wb2);
        v16bf bk2 = load_b_packed(wk + wb2);
        v16bf bv2 = load_b_packed(wv + wb2);
        cq = wmma_bf16(a, bq, cq);
        ck = wmma_bf16(a, bk, ck);
        cv = wmma_bf16(a, bv, cv);
        a = a2; bq = bq2; bk = bk2; bv = bv2;
    }
    cq = wmma_bf16(a, bq, cq);
    ck = wmma_bf16(a, bk, ck);
    cv = wmma_bf16(a, bv, cv);

    int n  = wave * 16 + (lane & 15);
    int mh = (lane >> 4) * 8;
    size_t bh = (size_t)b * HH + h;
#pragma unroll
    for (int r = 0; r < 8; ++r) {
        int s  = s0 + r + mh;
        int dk = n;
        qout[(bh * SS + s) * DKK + dk] = (__bf16)cq[r];
        {   // K packed: scores B-operand (K-dim=dk, N-dim=t=s)
            int l2 = (s & 15) + (((dk & 31) >> 4) << 4);
            size_t off = bh * ((SS/16) * 2 * 512)
                       + (((size_t)(s >> 4) * 2 + (dk >> 5)) << 9)
                       + (size_t)l2 * 16 + (dk & 15);
            kpk[off] = (__bf16)ck[r];
        }
        {   // V packed: P*V B-operand (K-dim=t=s, N-dim=dk)
            int l2 = (dk & 15) + (((s & 31) >> 4) << 4);
            size_t off = bh * ((SS/32) * 4 * 512)
                       + (((size_t)(s >> 5) * 4 + (dk >> 4)) << 9)
                       + (size_t)l2 * 16 + (s & 15);
            vpk[off] = (__bf16)cv[r];
        }
    }
}

// ------------------------------------------- attention: scores -> softmax -> P*V
__global__ void __launch_bounds__(128) k_attn(
    const __bf16* __restrict__ qb,    // [B,H,S,DK] bf16
    const __bf16* __restrict__ kpk,
    const __bf16* __restrict__ vpk,
    __bf16* __restrict__ attn)        // [B,S,H*DK] bf16
{
    extern __shared__ float smem[];
    float* sc     = smem;                    // 16*SS
    float* red    = smem + 16 * SS;          // 16*8
    float* rowmax = red + 16 * 8;            // 16
    float* rowsum = rowmax + 16;             // 16

    int st   = blockIdx.x & 127;
    int h    = (blockIdx.x >> 7) & 15;
    int b    = blockIdx.x >> 11;
    int wave = threadIdx.x >> 5;
    int lane = threadIdx.x & 31;
    int s0   = st * 16;
    size_t bh = (size_t)b * HH + h;

    const __bf16* qrow = qb + (bh * SS + s0) * DKK;
    v16bf qf0 = load_a_rowmajor(qrow, DKK, 0);
    v16bf qf1 = load_a_rowmajor(qrow, DKK, 32);

    // scores (double-buffered over t-tiles; wave w covers tt = w, w+4, ...)
    size_t kbase = bh * ((SS/16) * 2 * 512);
    v16bf kb0 = load_b_packed(kpk + kbase + (((size_t)wave * 2 + 0) << 9));
    v16bf kb1 = load_b_packed(kpk + kbase + (((size_t)wave * 2 + 1) << 9));
    for (int tt = wave; tt < SS / 16; tt += 4) {
        v16bf nb0, nb1;
        if (tt + 4 < SS / 16) {
            nb0 = load_b_packed(kpk + kbase + (((size_t)(tt + 4) * 2 + 0) << 9));
            nb1 = load_b_packed(kpk + kbase + (((size_t)(tt + 4) * 2 + 1) << 9));
        }
        v8f c = {};
        c = wmma_bf16(qf0, kb0, c);
        c = wmma_bf16(qf1, kb1, c);
        int n  = tt * 16 + (lane & 15);
        int mh = (lane >> 4) * 8;
#pragma unroll
        for (int r = 0; r < 8; ++r)
            sc[(size_t)(r + mh) * SS + n] = c[r] * 0.125f;   // 1/sqrt(64)
        kb0 = nb0; kb1 = nb1;
    }
    __syncthreads();

    // exact softmax over 16 rows (8 threads per row)
    int row = threadIdx.x >> 3;
    int seg = threadIdx.x & 7;
    {
        float m = -3.4e38f;
        const float* p = sc + (size_t)row * SS + seg * 256;
        for (int j = 0; j < 256; ++j) m = fmaxf(m, p[j]);
        red[row * 8 + seg] = m;
    }
    __syncthreads();
    if (seg == 0) {
        float m = red[row * 8];
        for (int j = 1; j < 8; ++j) m = fmaxf(m, red[row * 8 + j]);
        rowmax[row] = m;
    }
    __syncthreads();
    {
        float m = rowmax[row], s = 0.f;
        float* p = sc + (size_t)row * SS + seg * 256;
        for (int j = 0; j < 256; ++j) {
            float e = __expf(p[j] - m);
            p[j] = e;
            s += e;
        }
        red[row * 8 + seg] = s;
    }
    __syncthreads();
    if (seg == 0) {
        float s = 0.f;
        for (int j = 0; j < 8; ++j) s += red[row * 8 + j];
        rowsum[row] = s;
    }
    __syncthreads();

    // P * V (B double-buffered): wave w owns dk cols [16w,16w+16)
    size_t vbase = bh * ((SS/32) * 4 * 512);
    v8f c = {};
    int m  = lane & 15;
    int kb = (lane >> 4) * 16;
    v16bf bv = load_b_packed(vpk + vbase + ((size_t)wave << 9));
    for (int tb = 0; tb < SS / 32; ++tb) {
        v16bf bv2;
        if (tb + 1 < SS / 32)
            bv2 = load_b_packed(vpk + vbase + (((size_t)(tb + 1) * 4 + wave) << 9));
        Frag fa;
        const float* p = sc + (size_t)m * SS + tb * 32 + kb;
        f32x4 c0 = *(const f32x4*)(p);
        f32x4 c1 = *(const f32x4*)(p + 4);
        f32x4 c2 = *(const f32x4*)(p + 8);
        f32x4 c3 = *(const f32x4*)(p + 12);
#pragma unroll
        for (int j = 0; j < 4; ++j) {
            fa.u[j]      = (__bf16)c0[j];
            fa.u[4 + j]  = (__bf16)c1[j];
            fa.u[8 + j]  = (__bf16)c2[j];
            fa.u[12 + j] = (__bf16)c3[j];
        }
        c = wmma_bf16(fa.v, bv, c);
        bv = bv2;
    }
    int n  = wave * 16 + (lane & 15);
    int mh = (lane >> 4) * 8;
#pragma unroll
    for (int r = 0; r < 8; ++r) {
        int   mm = r + mh;
        float val = c[r] / rowsum[mm];
        attn[((size_t)b * SS + (s0 + mm)) * (HH * DKK) + h * DKK + n] = (__bf16)val;
    }
}

// ------------------------------------- out = x + attn_cat @ W2  (residual GEMM)
__global__ void __launch_bounds__(128) k_out(
    const __bf16* __restrict__ attn,   // [B*S, 1024] bf16
    const __bf16* __restrict__ w2,     // packed B-operand
    const float*  __restrict__ x,      // [B*S, 1024] f32
    float* __restrict__ out)           // [B*S, 1024] f32
{
    __shared__ __align__(16) __bf16 as[16 * DD];          // 32 KB

    int rb   = blockIdx.x;
    int cb   = blockIdx.y;
    int wave = threadIdx.x >> 5;
    int lane = threadIdx.x & 31;
    int s0   = rb * 16;
    int nt   = cb * 4 + wave;

    stage_strip_32k(as, attn + (size_t)s0 * DD);

    v8f c = {};
    v16bf a  = load_a_rowmajor(as, DD, 0);
    v16bf bw = load_b_packed(w2 + ((size_t)nt << 9));
    for (int kt = 0; kt < DD / 32 - 1; ++kt) {
        size_t wb2 = (((size_t)(kt + 1) * (DD/16) + nt) << 9);
        __builtin_prefetch(w2 + wb2 + 2048, 0, 0);
        v16bf a2  = load_a_rowmajor(as, DD, (kt + 1) * 32);
        v16bf bw2 = load_b_packed(w2 + wb2);
        c = wmma_bf16(a, bw, c);
        a = a2; bw = bw2;
    }
    c = wmma_bf16(a, bw, c);

    int n  = nt * 16 + (lane & 15);
    int mh = (lane >> 4) * 8;
#pragma unroll
    for (int r = 0; r < 8; ++r) {
        size_t off = (size_t)(s0 + r + mh) * DD + n;
        out[off] = c[r] + x[off];
    }
}

// ---------------------------------------------------------------------- launch
extern "C" void kernel_launch(void* const* d_in, const int* in_sizes, int n_in,
                              void* d_out, int out_size, void* d_ws, size_t ws_size,
                              hipStream_t stream) {
    const float* x   = (const float*)d_in[0];   // [B,S,D]
    const float* Wq  = (const float*)d_in[1];   // [H,DK,D]
    const float* Wk  = (const float*)d_in[2];
    const float* Wvd = (const float*)d_in[3];
    const float* Wvu = (const float*)d_in[4];   // [H,D,DK]
    float* out = (float*)d_out;

    const size_t N_X    = (size_t)BB * SS * DD;
    const size_t N_W    = (size_t)HH * DKK * DD;
    const size_t N_Q    = (size_t)BB * HH * SS * DKK;
    const size_t N_KPK  = (size_t)BB * HH * (SS/16) * 2 * 512;
    const size_t N_VPK  = (size_t)BB * HH * (SS/32) * 4 * 512;
    const size_t N_ATTN = (size_t)BB * SS * (HH * DKK);

    __bf16* p    = (__bf16*)d_ws;
    __bf16* Xbf  = p;  p += N_X;
    __bf16* Wqp  = p;  p += N_W;
    __bf16* Wkp  = p;  p += N_W;
    __bf16* Wvp  = p;  p += N_W;
    __bf16* W2p  = p;  p += N_W;
    __bf16* Qbf  = p;  p += N_Q;
    __bf16* Kpk  = p;  p += N_KPK;
    __bf16* Vpk  = p;  p += N_VPK;
    __bf16* Attn = p;  p += N_ATTN;
    (void)ws_size; (void)n_in; (void)in_sizes; (void)out_size;

    k_cast_bf16<<<(unsigned)((N_X + 255) / 256), 256, 0, stream>>>(x, Xbf, (int)N_X);
    k_pack_proj<<<(unsigned)((N_W + 255) / 256), 256, 0, stream>>>(Wq,  Wqp);
    k_pack_proj<<<(unsigned)((N_W + 255) / 256), 256, 0, stream>>>(Wk,  Wkp);
    k_pack_proj<<<(unsigned)((N_W + 255) / 256), 256, 0, stream>>>(Wvd, Wvp);
    k_pack_w2  <<<(unsigned)((N_W + 255) / 256), 256, 0, stream>>>(Wvu, W2p);

    k_qkv<<<BB * HH * (SS / 16), 128, 0, stream>>>(Xbf, Wqp, Wkp, Wvp, Qbf, Kpk, Vpk);

    const int smem_bytes = (16 * SS + 16 * 8 + 16 + 16) * (int)sizeof(float);
    (void)hipFuncSetAttribute((const void*)k_attn,
                              hipFuncAttributeMaxDynamicSharedMemorySize, smem_bytes);
    k_attn<<<BB * HH * (SS / 16), 128, smem_bytes, stream>>>(Qbf, Kpk, Vpk, Attn);

    k_out<<<dim3((BB * SS) / 16, DD / 64), 128, 0, stream>>>(Attn, W2p, x, out);
}
